// LearningNMS_72877005078724
// MI455X (gfx1250) — compile-verified, matrix operand order checked by reading, not run on hardware
//
#include <hip/hip_runtime.h>
#include <hip/hip_bf16.h>
#include <float.h>

typedef __attribute__((ext_vector_type(16))) __bf16 v16bf;
typedef __attribute__((ext_vector_type(8)))  float  v8f;

#define GG 512
#define HH 128
#define DP1 65
#define PDIM 136

// ---------------------------------------------------------------- utilities
__global__ void lnms_zero_kernel(float* p, int n) {
    int i = blockIdx.x * 256 + threadIdx.x;
    if (i < n) p[i] = 0.f;
}

// ---------------------------------------------------------------- geometry
__global__ __launch_bounds__(256) void lnms_geom_kernel(
    const float* __restrict__ boxes, float* __restrict__ geom,
    unsigned char* __restrict__ mask, float* __restrict__ gacc /*sum[6],ss[6]*/)
{
    __shared__ float sAcc[12];
    const int i = blockIdx.x, t = threadIdx.x;
    if (t < 12) sAcc[t] = 0.f;
    const float bx = boxes[i*4+0], by = boxes[i*4+1], bw = boxes[i*4+2], bh = boxes[i*4+3];
    const float x2i = bx + bw, y2i = by + bh, ai = bw*bh;
    const float cxi = bx + 0.5f*bw, cyi = by + 0.5f*bh;
    __syncthreads();
    float ls[6] = {0,0,0,0,0,0}, lss[6] = {0,0,0,0,0,0};
    for (int j = t; j < GG; j += 256) {
        float jx = boxes[j*4+0], jy = boxes[j*4+1], jw = boxes[j*4+2], jh = boxes[j*4+3];
        float ix1 = fmaxf(bx, jx), iy1 = fmaxf(by, jy);
        float ix2 = fminf(x2i, jx+jw), iy2 = fminf(y2i, jy+jh);
        float inter = fmaxf(ix2-ix1, 0.f) * fmaxf(iy2-iy1, 0.f);
        float aj = jw*jh;
        float iou = inter / (ai + aj - inter + 1e-6f);
        float g[6];
        g[0] = iou;
        g[1] = (cxi - (jx + 0.5f*jw)) / bw;
        g[2] = (cyi - (jy + 0.5f*jh)) / bh;
        g[3] = __logf(jw / bw);
        g[4] = __logf(jh / bh);
        g[5] = __logf(aj / ai);
        size_t base = ((size_t)i*GG + j)*6;
        #pragma unroll
        for (int c = 0; c < 6; ++c) {
            geom[base + c] = g[c];
            ls[c] += g[c]; lss[c] += g[c]*g[c];
        }
        mask[i*GG + j] = (iou > 0.3f) ? 1 : 0;
    }
    #pragma unroll
    for (int c = 0; c < 6; ++c) { atomicAdd(&sAcc[c], ls[c]); atomicAdd(&sAcc[6+c], lss[c]); }
    __syncthreads();
    if (t < 12) atomicAdd(&gacc[t], sAcc[t]);
}

// ---------------------------------------------------------------- f = [scores|feats]
__global__ void lnms_concat_kernel(const float* __restrict__ scores,
                                   const float* __restrict__ feats,
                                   float* __restrict__ fbuf)
{
    int idx = blockIdx.x * 256 + threadIdx.x;
    if (idx >= GG * DP1) return;
    int i = idx / DP1, c = idx % DP1;
    fbuf[idx] = (c == 0) ? scores[i] : feats[i*64 + (c-1)];
}

// ---------------------------------------------------------------- column stats of f
__global__ __launch_bounds__(128) void lnms_fstats_kernel(
    const float* __restrict__ fbuf, float* __restrict__ fmean, float* __restrict__ fvar)
{
    int c = threadIdx.x;
    if (c >= DP1) return;
    float s = 0.f, ss = 0.f;
    for (int r = 0; r < GG; ++r) { float v = fbuf[r*DP1 + c]; s += v; ss += v*v; }
    float m = s * (1.f/GG);
    fmean[c] = m;
    fvar[c]  = ss * (1.f/GG) - m*m;
}

// ---------------------------------------------------------------- fold BN0 into W0
__global__ __launch_bounds__(256) void lnms_fold_kernel(
    const float* __restrict__ gacc, const float* __restrict__ fmean, const float* __restrict__ fvar,
    const float* __restrict__ bn0g, const float* __restrict__ bn0b,
    const float* __restrict__ W0, const float* __restrict__ b0,
    float* __restrict__ W0gOut, float* __restrict__ b0fold,
    float* __restrict__ sFi, float* __restrict__ sFj)
{
    __shared__ float sS[PDIM], sT[PDIM];
    const int t = threadIdx.x;
    if (t < PDIM) {
        float m, v;
        if (t < 6)       { m = gacc[t]*(1.f/262144.f); v = gacc[6+t]*(1.f/262144.f) - m*m; }
        else if (t < 71) { m = fmean[t-6];  v = fvar[t-6];  }
        else             { m = fmean[t-71]; v = fvar[t-71]; }
        float sc = bn0g[t] * rsqrtf(v + 1e-3f);
        sS[t] = sc;
        sT[t] = bn0b[t] - m*sc;
    }
    __syncthreads();
    if (t < HH) {
        float acc = b0[t];
        for (int p = 0; p < PDIM; ++p) acc += sT[p]*W0[p*HH + t];
        b0fold[t] = acc;
        #pragma unroll
        for (int c = 0; c < 6; ++c) W0gOut[c*HH + t] = sS[c]*W0[c*HH + t];
    }
    if (t < DP1) { sFi[t] = sS[6 + t]; sFj[t] = sS[71 + t]; }
}

// ---------------------------------------------------------------- A = f@(s*W0_fi), B = f@(s*W0_fj)
__global__ __launch_bounds__(128) void lnms_ab_kernel(
    const float* __restrict__ fbuf, const float* __restrict__ sFi, const float* __restrict__ sFj,
    const float* __restrict__ W0, float* __restrict__ Abuf, float* __restrict__ Bbuf)
{
    __shared__ float sf[DP1], si[DP1], sj[DP1];
    const int i = blockIdx.x, t = threadIdx.x;
    if (t < DP1) { sf[t] = fbuf[i*DP1 + t]; si[t] = sFi[t]; sj[t] = sFj[t]; }
    __syncthreads();
    float a = 0.f, b = 0.f;
    for (int c = 0; c < DP1; ++c) {
        float fc = sf[c];
        a += fc * si[c] * W0[(6  + c)*HH + t];
        b += fc * sj[c] * W0[(71 + c)*HH + t];
    }
    Abuf[i*HH + t] = a;
    Bbuf[i*HH + t] = b;
}

// ---------------------------------------------------------------- W1^T -> bf16 (N-major)
__global__ void lnms_w1t_kernel(const float* __restrict__ W1, __bf16* __restrict__ W1T)
{
    int n = blockIdx.x, kk = threadIdx.x;
    W1T[n*HH + kk] = (__bf16)W1[kk*HH + n];
}

// ---------------------------------------------------------------- main pairwise WMMA kernel
// block: (i, jb) -> 128 pairs x 128 features. 8 waves; wave w = M-tile w.
__global__ __launch_bounds__(256) void lnms_pair_kernel(
    const float* __restrict__ geom, const float* __restrict__ Abuf, const float* __restrict__ Bbuf,
    const float* __restrict__ W0g, const float* __restrict__ b0fold,
    const __bf16* __restrict__ W1T, const float* __restrict__ b1,
    __bf16* __restrict__ h2out, float* __restrict__ bn1sum, float* __restrict__ bn1ss)
{
    __shared__ __bf16 sH1[128*128];     // h1 (bf16), later reused as h2 tile
    __shared__ float  sAi[128];
    __shared__ float  sGeom[128*6];
    __shared__ float  sWg[6*128];
    __shared__ float  sSum[128], sSS[128];

    const int i = blockIdx.x;
    const int jbase = blockIdx.y * 128;
    const int t = threadIdx.x;

    if (t < 128) { sAi[t] = Abuf[i*HH + t] + b0fold[t]; sSum[t] = 0.f; sSS[t] = 0.f; }
    {
        const float* gsrc = geom + ((size_t)i*GG + jbase)*6;
        for (int e = t; e < 128*6; e += 256) sGeom[e] = gsrc[e];
        for (int e = t; e < 6*128; e += 256) sWg[e] = W0g[e];
    }
    __syncthreads();

    // h1 = relu(geom@W0g' + A_i + B_j + b0'')  -> bf16 LDS, row-major [pair][h]
    for (int e = t; e < 128*128; e += 256) {
        int p = e >> 7, h = e & 127;
        float v = sAi[h] + Bbuf[(size_t)(jbase + p)*HH + h];
        #pragma unroll
        for (int c = 0; c < 6; ++c) v += sGeom[p*6 + c] * sWg[c*128 + h];
        sH1[e] = (__bf16)(v > 0.f ? v : 0.f);
    }
    __syncthreads();

    // WMMA phase: each wave owns rows [w*16, w*16+16)
    const int lane = t & 31;
    const int w    = t >> 5;
    const int m    = lane & 15;
    const int hi   = lane >> 4;
    const int row  = w*16 + m;

    // A fragments per ISA 16-bit A layout: lanes 0-15: K {0..7,16..23}+hi*8 offsets
    v16bf afrag[4];
    #pragma unroll
    for (int kc = 0; kc < 4; ++kc) {
        union { uint4 u[2]; v16bf v; } ua;
        ua.u[0] = *(const uint4*)(sH1 + row*128 + kc*32 +      hi*8);
        ua.u[1] = *(const uint4*)(sH1 + row*128 + kc*32 + 16 + hi*8);
        afrag[kc] = ua.v;
    }

    #pragma unroll
    for (int nt = 0; nt < 8; ++nt) {
        const int n = nt*16 + m;
        v8f acc = {};
        #pragma unroll
        for (int kc = 0; kc < 4; ++kc) {
            union { uint4 u[2]; v16bf v; } ub;
            ub.u[0] = *(const uint4*)(W1T + n*128 + kc*32 +      hi*8);
            ub.u[1] = *(const uint4*)(W1T + n*128 + kc*32 + 16 + hi*8);
            acc = __builtin_amdgcn_wmma_f32_16x16x32_bf16(
                      false, afrag[kc], false, ub.v, (short)0, acc, false, false);
        }
        // D layout: lane covers col n, rows w*16 + r + hi*8
        float bias = b1[n];
        #pragma unroll
        for (int r = 0; r < 8; ++r) {
            float v = acc[r] + bias;
            sH1[(w*16 + r + hi*8)*128 + n] = (__bf16)(v > 0.f ? v : 0.f);
        }
    }
    __syncthreads();

    // store h2 (bf16) + accumulate BN1 statistics
    {
        const int h  = t & 127;
        const int pr = t >> 7;
        float lsum = 0.f, lss = 0.f;
        const size_t base = ((size_t)i*GG + jbase)*HH;
        for (int e = 0; e < 64; ++e) {
            int p = pr + e*2;
            __bf16 hv = sH1[p*128 + h];
            float v = (float)hv;
            lsum += v; lss += v*v;
            h2out[base + (size_t)p*HH + h] = hv;
        }
        atomicAdd(&sSum[h], lsum);
        atomicAdd(&sSS[h],  lss);
    }
    __syncthreads();
    if (t < 128) { atomicAdd(&bn1sum[t], sSum[t]); atomicAdd(&bn1ss[t], sSS[t]); }
}

// ---------------------------------------------------------------- BN1 finalize
__global__ void lnms_bn1fin_kernel(const float* __restrict__ bn1sum, const float* __restrict__ bn1ss,
                                   const float* __restrict__ g, const float* __restrict__ b,
                                   float* __restrict__ scale1, float* __restrict__ shift1)
{
    int t = threadIdx.x;
    float m = bn1sum[t] * (1.f/262144.f);
    float v = bn1ss[t]  * (1.f/262144.f) - m*m;
    float sc = g[t] * rsqrtf(v + 1e-3f);
    scale1[t] = sc;
    shift1[t] = b[t] - m*sc;
}

// ---------------------------------------------------------------- masked max-pool + Wout
__global__ __launch_bounds__(128) void lnms_pool_kernel(
    const __bf16* __restrict__ h2, const unsigned char* __restrict__ mask,
    const float* __restrict__ scale1, const float* __restrict__ shift1,
    const float* __restrict__ Wout, const float* __restrict__ bout,
    float* __restrict__ fbuf)
{
    __shared__ float sPool[128];
    __shared__ unsigned char sMask[GG];
    const int i = blockIdx.x, t = threadIdx.x;
    for (int e = t; e < GG; e += 128) sMask[e] = mask[i*GG + e];
    __syncthreads();
    float mx = -FLT_MAX, mn = FLT_MAX;
    const __bf16* row = h2 + (size_t)i*GG*HH;
    for (int j = 0; j < GG; ++j) {
        if (sMask[j]) {
            float v = (float)row[(size_t)j*HH + t];
            mx = fmaxf(mx, v); mn = fminf(mn, v);
        }
    }
    float sc = scale1[t];
    sPool[t] = (sc >= 0.f ? mx : mn) * sc + shift1[t];   // affine commutes with masked max
    __syncthreads();
    if (t < DP1) {
        float acc = bout[t];
        for (int h = 0; h < HH; ++h) acc += sPool[h] * Wout[h*DP1 + t];
        fbuf[i*DP1 + t] = acc;
    }
}

// ---------------------------------------------------------------- head MLP + sigmoid
__global__ __launch_bounds__(128) void lnms_head_kernel(
    const float* __restrict__ fbuf,
    const float* __restrict__ Wf0, const float* __restrict__ bf0,
    const float* __restrict__ Wf1, const float* __restrict__ bf1,
    const float* __restrict__ Wsc, const float* __restrict__ bsc,
    float* __restrict__ out)
{
    __shared__ float sf[DP1], y0[128], y1[128];
    const int i = blockIdx.x, t = threadIdx.x;
    if (t < DP1) sf[t] = fbuf[i*DP1 + t];
    __syncthreads();
    float a = bf0[t];
    for (int c = 0; c < DP1; ++c) a += sf[c] * Wf0[c*128 + t];
    y0[t] = fmaxf(a, 0.f);
    __syncthreads();
    float b = bf1[t];
    for (int c = 0; c < 128; ++c) b += y0[c] * Wf1[c*128 + t];
    y1[t] = fmaxf(b, 0.f);
    __syncthreads();
    if (t == 0) {
        float z = bsc[0];
        for (int h = 0; h < 128; ++h) z += y1[h] * Wsc[h];
        out[i] = 1.f / (1.f + __expf(-z));
    }
}

// ================================================================ host launch
extern "C" void kernel_launch(void* const* d_in, const int* in_sizes, int n_in,
                              void* d_out, int out_size, void* d_ws, size_t ws_size,
                              hipStream_t stream)
{
    const float* boxes  = (const float*)d_in[0];
    const float* scores = (const float*)d_in[1];
    const float* feats  = (const float*)d_in[2];
    const float* bn0g   = (const float*)d_in[3];
    const float* bn0b   = (const float*)d_in[4];
    const float* W0     = (const float*)d_in[5];
    const float* b0     = (const float*)d_in[6];
    const float* W1     = (const float*)d_in[7];
    const float* b1     = (const float*)d_in[8];
    const float* bn1g   = (const float*)d_in[9];
    const float* bn1b   = (const float*)d_in[10];
    const float* Wout   = (const float*)d_in[11];
    const float* bout   = (const float*)d_in[12];
    const float* Wf0    = (const float*)d_in[13];
    const float* bf0    = (const float*)d_in[14];
    const float* Wf1    = (const float*)d_in[15];
    const float* bf1    = (const float*)d_in[16];
    const float* Wsc    = (const float*)d_in[17];
    const float* bsc    = (const float*)d_in[18];
    float* out = (float*)d_out;

    char* w = (char*)d_ws;
    size_t off = 0;
    auto take = [&](size_t bytes) { size_t o = off; off += (bytes + 255) & ~(size_t)255; return o; };

    float*         geomB  = (float*)        (w + take((size_t)GG*GG*6*4));
    unsigned char* maskB  = (unsigned char*)(w + take((size_t)GG*GG));
    __bf16*        h2B    = (__bf16*)       (w + take((size_t)GG*GG*HH*2));
    float*         Abuf   = (float*)        (w + take((size_t)GG*HH*4));
    float*         Bbuf   = (float*)        (w + take((size_t)GG*HH*4));
    float*         fbuf   = (float*)        (w + take((size_t)GG*DP1*4));
    __bf16*        W1T    = (__bf16*)       (w + take((size_t)HH*HH*2));
    float*         W0g    = (float*)        (w + take(6*HH*4));
    float*         b0fold = (float*)        (w + take(HH*4));
    float*         sFi    = (float*)        (w + take(DP1*4));
    float*         sFj    = (float*)        (w + take(DP1*4));
    float*         gacc   = (float*)        (w + take(12*4));
    float*         fmean  = (float*)        (w + take(DP1*4));
    float*         fvar   = (float*)        (w + take(DP1*4));
    float*         bn1acc = (float*)        (w + take(2*HH*4));
    float*         scale1 = (float*)        (w + take(HH*4));
    float*         shift1 = (float*)        (w + take(HH*4));
    (void)ws_size; (void)n_in; (void)in_sizes; (void)out_size;

    lnms_zero_kernel<<<1, 256, 0, stream>>>(gacc, 12);
    lnms_geom_kernel<<<GG, 256, 0, stream>>>(boxes, geomB, maskB, gacc);
    lnms_concat_kernel<<<(GG*DP1 + 255)/256, 256, 0, stream>>>(scores, feats, fbuf);

    for (int k = 0; k < 2; ++k) {
        lnms_zero_kernel<<<1, 256, 0, stream>>>(bn1acc, 2*HH);
        lnms_fstats_kernel<<<1, 128, 0, stream>>>(fbuf, fmean, fvar);
        lnms_fold_kernel<<<1, 256, 0, stream>>>(gacc, fmean, fvar,
                                                bn0g + k*PDIM, bn0b + k*PDIM,
                                                W0 + (size_t)k*PDIM*HH, b0 + k*HH,
                                                W0g, b0fold, sFi, sFj);
        lnms_ab_kernel<<<GG, 128, 0, stream>>>(fbuf, sFi, sFj, W0 + (size_t)k*PDIM*HH, Abuf, Bbuf);
        lnms_w1t_kernel<<<HH, HH, 0, stream>>>(W1 + (size_t)k*HH*HH, W1T);
        lnms_pair_kernel<<<dim3(GG, 4), 256, 0, stream>>>(geomB, Abuf, Bbuf, W0g, b0fold,
                                                          W1T, b1 + k*HH,
                                                          h2B, bn1acc, bn1acc + HH);
        lnms_bn1fin_kernel<<<1, 128, 0, stream>>>(bn1acc, bn1acc + HH,
                                                  bn1g + k*HH, bn1b + k*HH, scale1, shift1);
        lnms_pool_kernel<<<GG, 128, 0, stream>>>(h2B, maskB, scale1, shift1,
                                                 Wout + (size_t)k*HH*DP1, bout + k*DP1, fbuf);
    }

    lnms_head_kernel<<<GG, 128, 0, stream>>>(fbuf, Wf0, bf0, Wf1, bf1, Wsc, bsc, out);
}
